// PGNN_42992622633784
// MI455X (gfx1250) — matrix operand order repaired and checked by reading, not compiled, not run on hardware
//
#include <hip/hip_runtime.h>
#include <math.h>

// ---------------------------------------------------------------------------
// PGNN forward for MI455X (gfx1250, wave32).
//  - f16 WMMA 16x16x32, f32 accumulate; weights live in registers per wave.
//  - Gathers (x / h_agg rows, resident in 192MB L2) are double-buffered into
//    LDS with GLOBAL_LOAD_ASYNC_TO_LDS_B128 (ASYNCcnt) and overlapped with
//    the WMMA of the previous node.  Falls back to synchronous fill if the
//    async builtin is not declared by this toolchain.
// ---------------------------------------------------------------------------

typedef __attribute__((ext_vector_type(16))) _Float16 v16h;
typedef __attribute__((ext_vector_type(8)))  float    v8f;
typedef __attribute__((ext_vector_type(4)))  int      v4i;

#define NNODES 30000
#define KNBR   32
#define DIN    128
#define DHID   128
#define DOUT   64
#define CATW   256          // 2*DIN == 2*DHID
#define LDS_STRIDE  264     // f16 cat row stride: conflict-free b128 frag reads
#define RAW1_STRIDE 132     // f32 raw neighbor-row stride (128 + 4 pad)
#define RAW2_STRIDE 136     // f16 raw neighbor-row stride (128 + 8 pad)
#define NPB    15           // nodes per block -> grid of 2000

#if defined(__has_builtin)
#  if __has_builtin(__builtin_amdgcn_global_load_async_to_lds_b128)
#    define HAVE_ASYNC_LDS 1
#  endif
#endif
#ifndef HAVE_ASYNC_LDS
#  define HAVE_ASYNC_LDS 0
#endif

union AFrag { v16h v; uint4 u[2]; };
union H8    { uint4 u; _Float16 h[8]; };

#if HAVE_ASYNC_LDS
// Probe-learned signature: arg0 is a generic `v4i*`; pass the LDS side as an
// addrspace(3) v4i* (exact match if AS3-typed, implicit widening if generic).
typedef __attribute__((address_space(3))) v4i* lds_v4i_ptr;
__device__ __forceinline__ void async_b128(const void* g, void* l) {
  __builtin_amdgcn_global_load_async_to_lds_b128(
      (v4i*)g, (lds_v4i_ptr)l, 0, 0);
}
__device__ __forceinline__ void wait_async0() {
#if __has_builtin(__builtin_amdgcn_s_wait_asynccnt)
  __builtin_amdgcn_s_wait_asynccnt(0);
#else
  asm volatile("s_wait_asynccnt 0x0" ::: "memory");
#endif
}
#endif

// ---- shared helpers -------------------------------------------------------

// Stage one 32-float chunk of the 32x256 f16 concat matrix.
__device__ __forceinline__ void l1_fill(const float* nrow, const float* srow,
                                        float sc, _Float16* catRow, int c0)
{
  const float* src; float s;
  if (c0 < DIN) { src = nrow + c0;         s = sc;   }
  else          { src = srow + (c0 - DIN); s = 1.0f; }
  _Float16* dst = catRow + c0;
  #pragma unroll
  for (int j = 0; j < 8; ++j) {
    float4 v = ((const float4*)src)[j];
    dst[4*j+0] = (_Float16)(v.x * s);
    dst[4*j+1] = (_Float16)(v.y * s);
    dst[4*j+2] = (_Float16)(v.z * s);
    dst[4*j+3] = (_Float16)(v.w * s);
  }
}

__device__ __forceinline__ void l2_fill(const _Float16* nrow, const _Float16* srow,
                                        float sc, _Float16* catRow, int c0)
{
  const _Float16* src; float s;
  if (c0 < DHID) { src = nrow + c0;          s = sc;   }
  else           { src = srow + (c0 - DHID); s = 1.0f; }
  _Float16* dst = catRow + c0;
  #pragma unroll
  for (int j = 0; j < 4; ++j) {
    H8 a; a.u = ((const uint4*)src)[j];
    H8 b;
    #pragma unroll
    for (int e = 0; e < 8; ++e) b.h[e] = (_Float16)((float)a.h[e] * s);
    ((uint4*)dst)[j] = b.u;
  }
}

// Layer-1 per-wave compute: [32x256] x [256x16], relu, per-lane row sum.
__device__ __forceinline__ float l1_tile_compute(const _Float16* catA,
    const v16h B[8], float bias, int ln, int kb)
{
  v8f acc0, acc1;
  #pragma unroll
  for (int i = 0; i < 8; ++i) { acc0[i] = bias; acc1[i] = bias; }
  #pragma unroll
  for (int kt = 0; kt < 8; ++kt) {
    const _Float16* a0 = &catA[ ln       * LDS_STRIDE + kt * 32 + kb];
    const _Float16* a1 = &catA[(16 + ln) * LDS_STRIDE + kt * 32 + kb];
    AFrag A0, A1;
    A0.u[0] = *(const uint4*)(a0);      A0.u[1] = *(const uint4*)(a0 + 16);
    A1.u[0] = *(const uint4*)(a1);      A1.u[1] = *(const uint4*)(a1 + 16);
    acc0 = __builtin_amdgcn_wmma_f32_16x16x32_f16(
               false, A0.v, false, B[kt], (short)0, acc0, false, false);
    acc1 = __builtin_amdgcn_wmma_f32_16x16x32_f16(
               false, A1.v, false, B[kt], (short)0, acc1, false, false);
  }
  float s = 0.0f;
  #pragma unroll
  for (int r = 0; r < 8; ++r)
    s += fmaxf(acc0[r], 0.0f) + fmaxf(acc1[r], 0.0f);
  return s;
}

// Layer-2 per-wave compute: one 16x16 output tile.
__device__ __forceinline__ v8f l2_tile_compute(const _Float16* catA,
    const v16h B[8], float bias, int mrow, int kb)
{
  v8f acc;
  #pragma unroll
  for (int i = 0; i < 8; ++i) acc[i] = bias;
  #pragma unroll
  for (int kt = 0; kt < 8; ++kt) {
    const _Float16* a = &catA[mrow * LDS_STRIDE + kt * 32 + kb];
    AFrag A;
    A.u[0] = *(const uint4*)(a);        A.u[1] = *(const uint4*)(a + 16);
    acc = __builtin_amdgcn_wmma_f32_16x16x32_f16(
              false, A.v, false, B[kt], (short)0, acc, false, false);
  }
  return acc;
}

// Layer-2 epilogue: wp2 dot, deterministic reduce, normalize, linear head.
__device__ __forceinline__ void l2_epilogue(v8f acc, float wsc, float bp,
    const float* __restrict__ Wout, const float* __restrict__ bout,
    float* __restrict__ out, int n, float* posred, float* posbuf,
    int tid, int ln, int hb, int mi, int ni)
{
  #pragma unroll
  for (int r = 0; r < 8; ++r) {
    float v = fmaxf(acc[r], 0.0f) * wsc;
    v += __shfl_xor(v, 1, 32);
    v += __shfl_xor(v, 2, 32);
    v += __shfl_xor(v, 4, 32);
    v += __shfl_xor(v, 8, 32);
    if (ln == 0) posred[(mi * 16 + hb * 8 + r) * 4 + ni] = v;
  }
  __syncthreads();
  if (tid < 32) {
    float p = bp + posred[tid*4+0] + posred[tid*4+1]
                 + posred[tid*4+2] + posred[tid*4+3];
    float ss = p * p;
    ss += __shfl_xor(ss, 1, 32);
    ss += __shfl_xor(ss, 2, 32);
    ss += __shfl_xor(ss, 4, 32);
    ss += __shfl_xor(ss, 8, 32);
    ss += __shfl_xor(ss, 16, 32);
    float nrm = fmaxf(sqrtf(ss), 1e-12f);
    posbuf[tid] = p / nrm;
  }
  __syncthreads();
  if (tid < DOUT) {
    float a = bout[tid];
    const float* wr = Wout + tid * KNBR;
    #pragma unroll
    for (int k2 = 0; k2 < KNBR; ++k2) a += posbuf[k2] * wr[k2];
    out[(size_t)n * DOUT + tid] = a;
  }
  __syncthreads();
}

// ======================= Layer 1: x -> h_agg (f16) =========================
__global__ __launch_bounds__(256) void pgnn_layer1_wmma(
    const float* __restrict__ x, const float* __restrict__ dmax,
    const int* __restrict__ amax, const float* __restrict__ Wh1,
    const float* __restrict__ bh1, _Float16* __restrict__ hagg)
{
#if HAVE_ASYNC_LDS
  __shared__ float    rawN[2][KNBR * RAW1_STRIDE];
  __shared__ float    rawS[2][DIN];
  __shared__ _Float16 catA[2][32 * LDS_STRIDE];
#else
  __shared__ _Float16 catA[1][32 * LDS_STRIDE];
#endif

  const int tid  = threadIdx.x;
  const int lane = tid & 31;
  const int wave = tid >> 5;
  const int ln   = lane & 15;
  const int hb   = lane >> 4;
  const int kb   = hb * 8;
  const int row  = tid >> 3;          // fill row 0..31
  const int c0   = (tid & 7) * 32;    // fill column chunk

  // ---- B fragments: wave owns Wh1 output cols [16w, 16w+16) --------------
  const int ocol = wave * 16 + ln;
  v16h B[8];
  #pragma unroll
  for (int kt = 0; kt < 8; ++kt) {
    const float* wr = Wh1 + (size_t)ocol * CATW + kt * 32 + hb * 8;
    v16h b;
    #pragma unroll
    for (int j = 0; j < 16; ++j) b[j] = (_Float16)wr[j + (j & 8)];
    B[kt] = b;
  }
  const float bias = bh1[ocol];

  const int n0 = blockIdx.x * NPB;
  const int n1 = (n0 + NPB < NNODES) ? (n0 + NPB) : NNODES;

#if HAVE_ASYNC_LDS
  // Prologue: async gather of node n0 (neighbors + self) into buffer 0.
  {
    const int gidx = amax[n0 * KNBR + row];
    const float* g = x + (size_t)gidx * DIN + (tid & 7) * 16;
    float* l = &rawN[0][row * RAW1_STRIDE + (tid & 7) * 16];
    #pragma unroll
    for (int i = 0; i < 4; ++i) async_b128(g + 4*i, l + 4*i);
    if (tid < 32) async_b128(x + (size_t)n0 * DIN + tid * 4, &rawS[0][tid * 4]);
  }
  int buf = 0;
  for (int n = n0; n < n1; ++n) {
    wait_async0();
    __syncthreads();                                 // raw[buf] complete
    const float sc = dmax[n * KNBR + row];
    l1_fill(&rawN[buf][row * RAW1_STRIDE], rawS[buf], sc,
            &catA[buf][row * LDS_STRIDE], c0);
    if (n + 1 < n1) {                                // overlap next gather
      const int gidx = amax[(n + 1) * KNBR + row];
      const float* g = x + (size_t)gidx * DIN + (tid & 7) * 16;
      float* l = &rawN[buf ^ 1][row * RAW1_STRIDE + (tid & 7) * 16];
      #pragma unroll
      for (int i = 0; i < 4; ++i) async_b128(g + 4*i, l + 4*i);
      if (tid < 32)
        async_b128(x + (size_t)(n + 1) * DIN + tid * 4, &rawS[buf ^ 1][tid * 4]);
    }
    __syncthreads();                                 // catA[buf] ready
    float s = l1_tile_compute(catA[buf], B, bias, ln, kb);
    s += __shfl_xor(s, 16, 32);
    if (hb == 0)
      hagg[(size_t)n * DHID + ocol] = (_Float16)(s * (1.0f / 32.0f));
    buf ^= 1;
  }
#else
  for (int n = n0; n < n1; ++n) {
    const int gidx = amax[n * KNBR + row];
    const float sc = dmax[n * KNBR + row];
    l1_fill(x + (size_t)gidx * DIN, x + (size_t)n * DIN, sc,
            &catA[0][row * LDS_STRIDE], c0);
    __syncthreads();
    float s = l1_tile_compute(catA[0], B, bias, ln, kb);
    s += __shfl_xor(s, 16, 32);
    if (hb == 0)
      hagg[(size_t)n * DHID + ocol] = (_Float16)(s * (1.0f / 32.0f));
    __syncthreads();
  }
#endif
}

// ============ Layer 2 + normalize + linear_out: h_agg -> out ===============
__global__ __launch_bounds__(256) void pgnn_layer2_wmma(
    const _Float16* __restrict__ hagg, const float* __restrict__ dmax,
    const int* __restrict__ amax, const float* __restrict__ Wh2,
    const float* __restrict__ bh2, const float* __restrict__ wp2,
    const float* __restrict__ bp2, const float* __restrict__ Wout,
    const float* __restrict__ bout, float* __restrict__ out)
{
#if HAVE_ASYNC_LDS
  __shared__ _Float16 rawN[2][KNBR * RAW2_STRIDE];
  __shared__ _Float16 rawS[2][DHID];
  __shared__ _Float16 catA[2][32 * LDS_STRIDE];
#else
  __shared__ _Float16 catA[1][32 * LDS_STRIDE];
#endif
  __shared__ float posred[32 * 4];
  __shared__ float posbuf[32];

  const int tid  = threadIdx.x;
  const int lane = tid & 31;
  const int wave = tid >> 5;
  const int ln   = lane & 15;
  const int hb   = lane >> 4;
  const int kb   = hb * 8;
  const int mi   = wave >> 2;
  const int ni   = wave & 3;
  const int row  = tid >> 3;
  const int c0   = (tid & 7) * 32;

  const int ocol = ni * 16 + ln;      // 0..63
  v16h B[8];
  #pragma unroll
  for (int kt = 0; kt < 8; ++kt) {
    const float* wr = Wh2 + (size_t)ocol * CATW + kt * 32 + hb * 8;
    v16h b;
    #pragma unroll
    for (int j = 0; j < 16; ++j) b[j] = (_Float16)wr[j + (j & 8)];
    B[kt] = b;
  }
  const float bias = bh2[ocol];
  const float wsc  = wp2[ocol];
  const float bp   = bp2[0];

  const int n0 = blockIdx.x * NPB;
  const int n1 = (n0 + NPB < NNODES) ? (n0 + NPB) : NNODES;

#if HAVE_ASYNC_LDS
  {
    const int gidx = amax[n0 * KNBR + row];
    const _Float16* g = hagg + (size_t)gidx * DHID + (tid & 7) * 16;
    _Float16* l = &rawN[0][row * RAW2_STRIDE + (tid & 7) * 16];
    #pragma unroll
    for (int i = 0; i < 2; ++i) async_b128(g + 8*i, l + 8*i);
    if (tid < 16) async_b128(hagg + (size_t)n0 * DHID + tid * 8, &rawS[0][tid * 8]);
  }
  int buf = 0;
  for (int n = n0; n < n1; ++n) {
    wait_async0();
    __syncthreads();
    const float sc = dmax[n * KNBR + row];
    l2_fill(&rawN[buf][row * RAW2_STRIDE], rawS[buf], sc,
            &catA[buf][row * LDS_STRIDE], c0);
    if (n + 1 < n1) {
      const int gidx = amax[(n + 1) * KNBR + row];
      const _Float16* g = hagg + (size_t)gidx * DHID + (tid & 7) * 16;
      _Float16* l = &rawN[buf ^ 1][row * RAW2_STRIDE + (tid & 7) * 16];
      #pragma unroll
      for (int i = 0; i < 2; ++i) async_b128(g + 8*i, l + 8*i);
      if (tid < 16)
        async_b128(hagg + (size_t)(n + 1) * DHID + tid * 8, &rawS[buf ^ 1][tid * 8]);
    }
    __syncthreads();
    v8f acc = l2_tile_compute(catA[buf], B, bias, mi * 16 + ln, kb);
    l2_epilogue(acc, wsc, bp, Wout, bout, out, n,
                posred, posbuf, tid, ln, hb, mi, ni);
    buf ^= 1;
  }
#else
  for (int n = n0; n < n1; ++n) {
    const int gidx = amax[n * KNBR + row];
    const float sc = dmax[n * KNBR + row];
    l2_fill(hagg + (size_t)gidx * DHID, hagg + (size_t)n * DHID, sc,
            &catA[0][row * LDS_STRIDE], c0);
    __syncthreads();
    v8f acc = l2_tile_compute(catA[0], B, bias, mi * 16 + ln, kb);
    l2_epilogue(acc, wsc, bp, Wout, bout, out, n,
                posred, posbuf, tid, ln, hb, mi, ni);
  }
#endif
}

// ===========================================================================
extern "C" void kernel_launch(void* const* d_in, const int* in_sizes, int n_in,
                              void* d_out, int out_size, void* d_ws, size_t ws_size,
                              hipStream_t stream)
{
  const float* x    = (const float*)d_in[0];
  const float* dmax = (const float*)d_in[1];
  const int*   amax = (const int*)  d_in[2];
  const float* Wh1  = (const float*)d_in[3];
  const float* bh1  = (const float*)d_in[4];
  // d_in[5] = wp1, d_in[6] = bp1: layer-1 position head is discarded upstream
  const float* Wh2  = (const float*)d_in[7];
  const float* bh2  = (const float*)d_in[8];
  const float* wp2  = (const float*)d_in[9];
  const float* bp2  = (const float*)d_in[10];
  const float* Wout = (const float*)d_in[11];
  const float* bout = (const float*)d_in[12];
  float* out = (float*)d_out;

  _Float16* hagg = (_Float16*)d_ws;  // 30000*128 f16 = 7.68 MB scratch

  dim3 grid((NNODES + NPB - 1) / NPB), block(256);
  pgnn_layer1_wmma<<<grid, block, 0, stream>>>(x, dmax, amax, Wh1, bh1, hagg);
  pgnn_layer2_wmma<<<grid, block, 0, stream>>>(hagg, dmax, amax, Wh2, bh2,
                                               wp2, bp2, Wout, bout, out);
  (void)in_sizes; (void)n_in; (void)out_size; (void)ws_size;
}
